// SLICSegmentation_87514253623385
// MI455X (gfx1250) — compile-verified, matrix operand order checked by reading, not run on hardware
//
#include <hip/hip_runtime.h>
#include <hip/hip_bf16.h>
#include <cfloat>
#include <climits>

#define HEIGHT   224
#define WIDTH    224
#define HW       (HEIGHT * WIDTH)
#define NCLUST   196
#define NPAD     208           // 13 tiles of 16 clusters
#define NTILES   13
#define NBHD     10
#define MAX_ITER 50
#define BATCH    4
#define MW       1.25f         // m / S = 20 / 16
#define PIX_PER_BLOCK   128    // 8 waves * 16 pixels
#define BLOCKS_PER_BATCH (HW / PIX_PER_BLOCK)  // 392

typedef __attribute__((ext_vector_type(16))) _Float16     v16h;
typedef __attribute__((ext_vector_type(8)))  float        v8f;
typedef __attribute__((ext_vector_type(4)))  unsigned int v4u;

static __device__ __forceinline__ unsigned int pack_h2(_Float16 a, _Float16 b) {
  return (unsigned int)__builtin_bit_cast(unsigned short, a) |
         ((unsigned int)__builtin_bit_cast(unsigned short, b) << 16);
}

// ---------------------------------------------------------------------------
// Stage 1: sequential nearest-minima seeding (one block per batch image).
// Matches the jax.lax.scan over clusters with an occupancy map.
// ---------------------------------------------------------------------------
__global__ void __launch_bounds__(256)
slic_seed_kernel(const float* __restrict__ grad, float* __restrict__ cents_out) {
  const int b   = blockIdx.x;
  const int tid = threadIdx.x;
  __shared__ unsigned int occ[HW / 32];   // occupancy bitmap (6272 B)
  __shared__ float redV[256];
  __shared__ int   redI[256];

  for (int i = tid; i < HW / 32; i += 256) occ[i] = 0u;
  __syncthreads();

  const float* g = grad + (size_t)b * HW;

  for (int c = 0; c < NCLUST; ++c) {
    // grid centroid: num_cols = num_rows = 14, cell = 16 -> (8+16i, 8+16j)
    const int gy = 8 + 16 * (c / 14);
    const int gx = 8 + 16 * (c % 14);
    const int r0 = max(0, gy - NBHD), r1 = min(HEIGHT, gy + NBHD);
    const int c0 = max(0, gx - NBHD), c1 = min(WIDTH,  gx + NBHD);
    const int wh = r1 - r0, ww = c1 - c0, wn = wh * ww;

    // pass 1: window minimum of gradient
    float lmin = FLT_MAX;
    for (int i = tid; i < wn; i += 256) {
      const int yy = r0 + i / ww, xx = c0 + i % ww;
      lmin = fminf(lmin, g[yy * WIDTH + xx]);
    }
    redV[tid] = lmin;
    __syncthreads();
    for (int s = 128; s > 0; s >>= 1) {
      if (tid < s) redV[tid] = fminf(redV[tid], redV[tid + s]);
      __syncthreads();
    }
    const float mv = redV[0];
    __syncthreads();

    // pass 2: first (lowest flat index) unoccupied pixel equal to mv
    int li = INT_MAX;
    for (int i = tid; i < wn; i += 256) {
      const int yy = r0 + i / ww, xx = c0 + i % ww;
      const int flat = yy * WIDTH + xx;
      const bool occb = (occ[flat >> 5] >> (flat & 31)) & 1u;
      if (g[flat] == mv && !occb) li = min(li, flat);
    }
    redI[tid] = li;
    __syncthreads();
    for (int s = 128; s > 0; s >>= 1) {
      if (tid < s) redI[tid] = min(redI[tid], redI[tid + s]);
      __syncthreads();
    }
    if (tid == 0) {
      const int idx = redI[0];
      float cy = (float)gy, cx = (float)gx;
      if (idx != INT_MAX) {
        occ[idx >> 5] |= (1u << (idx & 31));
        cy = (float)(idx / WIDTH);
        cx = (float)(idx % WIDTH);
      }
      cents_out[((size_t)b * NCLUST + c) * 2 + 0] = cy;
      cents_out[((size_t)b * NCLUST + c) * 2 + 1] = cx;
    }
    __syncthreads();   // publish occ update before next cluster
  }
}

// ---------------------------------------------------------------------------
// Stage 2 init: yc/xc/colors from seeded centroids; zero accumulators.
// ---------------------------------------------------------------------------
__global__ void slic_init_kernel(const float* __restrict__ x,
                                 const float* __restrict__ cents,
                                 int* __restrict__ yc, int* __restrict__ xc,
                                 float* __restrict__ colors,
                                 float* __restrict__ accum) {
  const int t = blockIdx.x * blockDim.x + threadIdx.x;
  if (t >= BATCH * NCLUST) return;
  const int b = t / NCLUST;
  const float cy = cents[t * 2 + 0], cx = cents[t * 2 + 1];
  const int iy = min(max((int)rintf(cy), 0), HEIGHT - 1);
  const int ix = min(max((int)rintf(cx), 0), WIDTH - 1);
  yc[t] = iy; xc[t] = ix;
  const int pix = iy * WIDTH + ix;
  colors[t * 3 + 0] = x[((size_t)b * 3 + 0) * HW + pix];
  colors[t * 3 + 1] = x[((size_t)b * 3 + 1) * HW + pix];
  colors[t * 3 + 2] = x[((size_t)b * 3 + 2) * HW + pix];
#pragma unroll
  for (int k = 0; k < 6; ++k) accum[t * 6 + k] = 0.f;
}

// ---------------------------------------------------------------------------
// Assignment: per-wave 16-pixel x 196-cluster distance argmin via WMMA.
// score(p,c) = f_p . (-2 g_c)  +  |g_c|^2      (|f_p|^2 dropped: row-constant)
// ---------------------------------------------------------------------------

// ds_swizzle group-of-32 immediates: xor_mask<<10 | and_mask(0x1f)
#define SWZ_X1 0x041f
#define SWZ_X2 0x081f
#define SWZ_X4 0x101f
#define SWZ_X8 0x201f

#define REDUCE_STEP(IMM)                                                       \
  _Pragma("unroll")                                                            \
  for (int r = 0; r < 8; ++r) {                                                \
    const float ov = __builtin_bit_cast(                                       \
        float, __builtin_amdgcn_ds_swizzle(                                    \
                   __builtin_bit_cast(int, bestV[r]), (IMM)));                 \
    const int oi = __builtin_amdgcn_ds_swizzle(bestI[r], (IMM));               \
    if (ov < bestV[r] || (ov == bestV[r] && oi < bestI[r])) {                  \
      bestV[r] = ov; bestI[r] = oi;                                            \
    }                                                                          \
  }

__global__ void __launch_bounds__(256)
slic_assign_kernel(const float* __restrict__ x,
                   const int* __restrict__ yc, const int* __restrict__ xc,
                   const float* __restrict__ colors,
                   int* __restrict__ labels, float* __restrict__ accum) {
  // Row NPAD is an all-zero row read by lanes 16..31 (keeps EXEC uniform).
  __shared__ __attribute__((aligned(16))) unsigned int sB[NPAD + 1][4];
  __shared__ float sBias[NPAD];

  const int b     = blockIdx.x / BLOCKS_PER_BATCH;
  const int pbase = (blockIdx.x % BLOCKS_PER_BATCH) * PIX_PER_BLOCK;
  const int tid   = threadIdx.x;
  const float msq = MW * MW;

  // Stage cluster features (-2*g packed f16x2, |g|^2) in LDS.
  if (tid <= NPAD) {
    if (tid < NCLUST) {
      const int t = b * NCLUST + tid;
      const float c0 = colors[t * 3 + 0];
      const float c1 = colors[t * 3 + 1];
      const float c2 = colors[t * 3 + 2];
      const float fy = (float)yc[t], fx = (float)xc[t];
      sB[tid][0] = pack_h2((_Float16)(-2.f * c0), (_Float16)(-2.f * c1));
      sB[tid][1] = pack_h2((_Float16)(-2.f * c2), (_Float16)(-2.f * MW * fy));
      sB[tid][2] = pack_h2((_Float16)(-2.f * MW * fx), (_Float16)0.f);
      sB[tid][3] = 0u;
      sBias[tid] = c0 * c0 + c1 * c1 + c2 * c2 + msq * (fy * fy + fx * fx);
    } else {
#pragma unroll
      for (int k = 0; k < 4; ++k) sB[tid][k] = 0u;
      if (tid < NPAD) sBias[tid] = 1e30f;  // padded clusters never win
    }
  }
  __syncthreads();

  const int wave = tid >> 5;
  const int lane = tid & 31;
  const int p    = pbase + wave * 16;

  // Pull this block's pixel rows toward the WGP (global_prefetch_b8).
  __builtin_prefetch(&x[((size_t)b * 3 + 0) * HW + pbase + tid], 0, 1);

  // A fragment (16x32 f16): lanes 0-15 hold K=0..7 (elems 0-7) and K=16..23
  // (elems 8-15) of row M=lane; only K=0..4 are non-zero. Lanes 16-31 zero.
  v16h a = {};
  if (lane < 16) {
    const int pix = p + lane;
    const int py = pix / WIDTH, px = pix % WIDTH;
    a[0] = (_Float16)x[((size_t)b * 3 + 0) * HW + pix];
    a[1] = (_Float16)x[((size_t)b * 3 + 1) * HW + pix];
    a[2] = (_Float16)x[((size_t)b * 3 + 2) * HW + pix];
    a[3] = (_Float16)(MW * (float)py);
    a[4] = (_Float16)(MW * (float)px);
  }

  float bestV[8];
  int   bestI[8];
#pragma unroll
  for (int r = 0; r < 8; ++r) { bestV[r] = FLT_MAX; bestI[r] = 0; }

  const int col = lane & 15;   // output column N for this lane (both halves)

#pragma unroll
  for (int t = 0; t < NTILES; ++t) {
    // B fragment (32x16 f16): lanes 0-15 hold column N=lane, K=0..7 via one
    // ds_load_b128; lanes 16-31 read the zero row (K=8..31 are zero-padded).
    const int brow = (lane < 16) ? (t * 16 + lane) : NPAD;
    v16h bf = {};
    *(v4u*)&bf = *(const v4u*)sB[brow];

    v8f acc = {};
    acc = __builtin_amdgcn_wmma_f32_16x16x32_f16(
        /*neg_a=*/false, a, /*neg_b=*/false, bf,
        /*c_mod=*/(short)0, acc, /*reuse_a=*/false, /*reuse_b=*/false);

    const int   cid  = t * 16 + col;
    const float bias = sBias[cid];
#pragma unroll
    for (int r = 0; r < 8; ++r) {
      const float v = acc[r] + bias;
      if (v < bestV[r]) { bestV[r] = v; bestI[r] = cid; }
    }
  }

  // Cross-lane min+argmin inside each 16-lane half via ds_swizzle butterflies.
  REDUCE_STEP(SWZ_X1)
  REDUCE_STEP(SWZ_X2)
  REDUCE_STEP(SWZ_X4)
  REDUCE_STEP(SWZ_X8)

  // Lane 0 owns rows M=0..7 (pixels p..p+7), lane 16 rows M=8..15.
  if (col == 0) {
    const int rowBase = (lane == 0) ? 0 : 8;
    const int pix0 = p + rowBase;
    int4* lp = (int4*)(labels + (size_t)b * HW + pix0);
    lp[0] = make_int4(bestI[0], bestI[1], bestI[2], bestI[3]);
    lp[1] = make_int4(bestI[4], bestI[5], bestI[6], bestI[7]);
#pragma unroll
    for (int r = 0; r < 8; ++r) {
      const int pix = pix0 + r;
      const int py = pix / WIDTH, px = pix % WIDTH;
      float* acc = accum + ((size_t)b * NCLUST + bestI[r]) * 6;
      atomicAdd(acc + 0, 1.0f);
      atomicAdd(acc + 1, (float)py);
      atomicAdd(acc + 2, (float)px);
      atomicAdd(acc + 3, x[((size_t)b * 3 + 0) * HW + pix]);
      atomicAdd(acc + 4, x[((size_t)b * 3 + 1) * HW + pix]);
      atomicAdd(acc + 5, x[((size_t)b * 3 + 2) * HW + pix]);
    }
  }
}

// ---------------------------------------------------------------------------
// Centroid update + accumulator re-zero for the next iteration.
// ---------------------------------------------------------------------------
__global__ void slic_update_kernel(int* __restrict__ yc, int* __restrict__ xc,
                                   float* __restrict__ colors,
                                   float* __restrict__ accum) {
  const int t = blockIdx.x * blockDim.x + threadIdx.x;
  if (t >= BATCH * NCLUST) return;
  const float cnt = accum[t * 6 + 0];
  const float sy  = accum[t * 6 + 1];
  const float sx  = accum[t * 6 + 2];
  const float s0  = accum[t * 6 + 3];
  const float s1  = accum[t * 6 + 4];
  const float s2  = accum[t * 6 + 5];
  if (cnt > 0.f) {
    const float safe = fmaxf(cnt, 1.f);
    yc[t] = min(max((int)rintf(sy / safe), 0), HEIGHT - 1);
    xc[t] = min(max((int)rintf(sx / safe), 0), WIDTH - 1);
    colors[t * 3 + 0] = s0 / safe;
    colors[t * 3 + 1] = s1 / safe;
    colors[t * 3 + 2] = s2 / safe;
  }
#pragma unroll
  for (int k = 0; k < 6; ++k) accum[t * 6 + k] = 0.f;
}

// ---------------------------------------------------------------------------
extern "C" void kernel_launch(void* const* d_in, const int* in_sizes, int n_in,
                              void* d_out, int out_size, void* d_ws, size_t ws_size,
                              hipStream_t stream) {
  (void)in_sizes; (void)n_in; (void)out_size; (void)ws_size;
  const float* x    = (const float*)d_in[0];   // (4,3,224,224) f32
  const float* grad = (const float*)d_in[1];   // (4,224,224)   f32

  float* cents  = (float*)d_out;                          // (4,196,2) f32
  int*   labels = ((int*)d_out) + BATCH * NCLUST * 2;     // (4,224,224) i32

  char* ws = (char*)d_ws;
  int*   yc     = (int*)ws;    ws += BATCH * NCLUST * sizeof(int);
  int*   xc     = (int*)ws;    ws += BATCH * NCLUST * sizeof(int);
  float* colors = (float*)ws;  ws += BATCH * NCLUST * 3 * sizeof(float);
  float* accum  = (float*)ws;  // BATCH*NCLUST*6 floats

  slic_seed_kernel<<<BATCH, 256, 0, stream>>>(grad, cents);
  slic_init_kernel<<<(BATCH * NCLUST + 255) / 256, 256, 0, stream>>>(
      x, cents, yc, xc, colors, accum);
  for (int it = 0; it < MAX_ITER; ++it) {
    slic_assign_kernel<<<BATCH * BLOCKS_PER_BATCH, 256, 0, stream>>>(
        x, yc, xc, colors, labels, accum);
    slic_update_kernel<<<(BATCH * NCLUST + 255) / 256, 256, 0, stream>>>(
        yc, xc, colors, accum);
  }
}
